// ODENet_2929167696200
// MI455X (gfx1250) — compile-verified, matrix operand order checked by reading, not run on hardware
//
#include <hip/hip_runtime.h>

// ---------------------------------------------------------------------------
// Types for CDNA5 WMMA (wave32, v_wmma_f32_16x16x32_bf16)
// ---------------------------------------------------------------------------
typedef __attribute__((ext_vector_type(16))) __bf16 bf16x16;
typedef __attribute__((ext_vector_type(8)))  __bf16 bf16x8;
typedef __attribute__((ext_vector_type(8)))  float  f32x8;
typedef __attribute__((address_space(3)))    __bf16 lds_bf16;

union FragU { bf16x16 v; bf16x8 h[2]; };

static __device__ __forceinline__ __bf16 f2bf(float f) {
    unsigned u = __builtin_bit_cast(unsigned, f);
    unsigned r = u + 0x7FFFu + ((u >> 16) & 1u);   // round-to-nearest-even
    unsigned short s = (unsigned short)(r >> 16);
    return __builtin_bit_cast(__bf16, s);
}

// CDNA5 async copy: 16B global -> LDS, tracked with ASYNCcnt (no VGPR staging)
static __device__ __forceinline__ void async_ld16(const __bf16* g, __bf16* ldsp) {
    unsigned loff = (unsigned)(unsigned long long)(lds_bf16*)ldsp;
    asm volatile("global_load_async_to_lds_b128 %0, %1, off"
                 :: "v"(loff), "v"(g)
                 : "memory");
}
static __device__ __forceinline__ void wait_async0() {
    asm volatile("s_wait_asynccnt 0x0" ::: "memory");
}

// ---------------------------------------------------------------------------
// TN bf16 GEMM:  C[M,N] = act( A[M,K] @ B[N,K]^T + bias[N] )
//   RELU_BF16_OUT == 1 : bf16 output with ReLU (hidden activations)
//   RELU_BF16_OUT == 0 : f32 output, no ReLU (k-stage output)
// Block tile 128 x (NWJ*64), BK=64, 256 threads = 8 waves (2 along M x 4 along N),
// wave tile 64 x (NWJ*16). Double-buffered LDS filled by async-to-LDS copies;
// 8*NWJ WMMAs between barriers.
// ---------------------------------------------------------------------------
#define BK  64
#define LDK 72   // padded LDS row stride (bf16 elems): 144B, 16B-aligned, no conflicts

template<int RELU_BF16_OUT, int NWJ>   // NWJ: 16-wide wmma tiles per wave along N
__global__ __launch_bounds__(256) void gemm_bf16_tn(
    const __bf16* __restrict__ A,     // M x K, row-major
    const __bf16* __restrict__ B,     // N x K, row-major
    const float*  __restrict__ bias,  // N
    void*         __restrict__ Cout,  // M x N
    int M, int N, int K)
{
    constexpr int BM     = 128;
    constexpr int BN     = NWJ * 16 * 4;            // 128 or 256
    constexpr int CPR    = BK / 8;                  // 16B chunks per tile row
    constexpr int ACHUNK = (BM * BK) / (8 * 256);   // 16B chunks per thread (A)
    constexpr int BCHUNK = (BN * BK) / (8 * 256);   // 16B chunks per thread (B)
    constexpr int TILEE  = (BM + BN) * LDK;         // bf16 elems per buffer

    __shared__ __bf16 smem[2 * TILEE];

    const int tid  = threadIdx.x;
    const int lane = tid & 31;
    const int wid  = tid >> 5;
    const int wy   = wid & 1;          // 2 waves along M
    const int wx   = wid >> 1;         // 4 waves along N
    const int bm0  = blockIdx.x * BM;
    const int bn0  = blockIdx.y * BN;

    const int half = lane >> 4;        // 0: K chunks {0..7,16..23}, 1: {8..15,24..31}
    const int mlan = lane & 15;
    const int coff = half * 8;

    f32x8 acc[4][NWJ];
#pragma unroll
    for (int i = 0; i < 4; ++i)
#pragma unroll
        for (int j = 0; j < NWJ; ++j)
#pragma unroll
            for (int e = 0; e < 8; ++e) acc[i][j][e] = 0.0f;

    // Per-thread async copy slots: chunk c -> row c/CPR, col (c%CPR)*8 of the tile
    const __bf16* ga[ACHUNK]; unsigned la[ACHUNK];
#pragma unroll
    for (int i = 0; i < ACHUNK; ++i) {
        const int c = tid + 256 * i, r = c / CPR, cc = (c % CPR) * 8;
        ga[i] = A + (size_t)(bm0 + r) * K + cc;
        la[i] = r * LDK + cc;
    }
    const __bf16* gb[BCHUNK]; unsigned lb[BCHUNK];
#pragma unroll
    for (int i = 0; i < BCHUNK; ++i) {
        const int c = tid + 256 * i, r = c / CPR, cc = (c % CPR) * 8;
        gb[i] = B + (size_t)(bn0 + r) * K + cc;
        lb[i] = BM * LDK + r * LDK + cc;
    }

    auto issue_tile = [&](int buf, int k0) {
        __bf16* base = smem + buf * TILEE;
#pragma unroll
        for (int i = 0; i < ACHUNK; ++i) async_ld16(ga[i] + k0, base + la[i]);
#pragma unroll
        for (int i = 0; i < BCHUNK; ++i) async_ld16(gb[i] + k0, base + lb[i]);
    };

    const int ksteps = K / BK;
    issue_tile(0, 0);

    for (int ks = 0; ks < ksteps; ++ks) {
        const int buf = ks & 1;
        wait_async0();          // tile ks landed in LDS
        __syncthreads();        // visible to all waves; prev buf free for overwrite
        if (ks + 1 < ksteps) issue_tile(buf ^ 1, (ks + 1) * BK);

        const __bf16* sA = smem + buf * TILEE;
        const __bf16* sB = sA + BM * LDK;

#pragma unroll
        for (int kk = 0; kk < BK; kk += 32) {
            FragU af[4], bfr[NWJ];
#pragma unroll
            for (int i = 0; i < 4; ++i) {
                const __bf16* p = &sA[(wy * 64 + i * 16 + mlan) * LDK + kk + coff];
                af[i].h[0] = *(const bf16x8*)(p);
                af[i].h[1] = *(const bf16x8*)(p + 16);
            }
#pragma unroll
            for (int j = 0; j < NWJ; ++j) {
                const __bf16* p = &sB[(wx * (NWJ * 16) + j * 16 + mlan) * LDK + kk + coff];
                bfr[j].h[0] = *(const bf16x8*)(p);
                bfr[j].h[1] = *(const bf16x8*)(p + 16);
            }
#pragma unroll
            for (int i = 0; i < 4; ++i)
#pragma unroll
                for (int j = 0; j < NWJ; ++j)
                    acc[i][j] = __builtin_amdgcn_wmma_f32_16x16x32_bf16(
                        false, af[i].v, false, bfr[j].v,
                        (short)0, acc[i][j], false, false);
        }
    }

    // Epilogue. C/D layout: lane L -> N = L&15; VGPR r -> M = r + (L>=16 ? 8 : 0)
    const int nl  = lane & 15;
    const int mr0 = half * 8;
#pragma unroll
    for (int i = 0; i < 4; ++i) {
#pragma unroll
        for (int j = 0; j < NWJ; ++j) {
            const int gn = bn0 + wx * (NWJ * 16) + j * 16 + nl;
            const int gm = bm0 + wy * 64 + i * 16 + mr0;
            const float bz = bias[gn];
#pragma unroll
            for (int r = 0; r < 8; ++r) {
                float v = acc[i][j][r] + bz;
                if (RELU_BF16_OUT) {
                    v = v > 0.0f ? v : 0.0f;
                    ((__bf16*)Cout)[(size_t)(gm + r) * N + gn] = f2bf(v);
                } else {
                    ((float*)Cout)[(size_t)(gm + r) * N + gn] = v;
                }
            }
        }
    }
}

// ---------------------------------------------------------------------------
// Elementwise helpers (f32 residual stream, bf16 GEMM operands)
// ---------------------------------------------------------------------------
__global__ void prep_stage(const float* __restrict__ y,
                           const float* __restrict__ t1, const float* __restrict__ t2,
                           const float* __restrict__ t3, const float* __restrict__ t4,
                           const float* __restrict__ t5,
                           float c1, float c2, float c3, float c4, float c5,
                           int nt, __bf16* __restrict__ out, int n)
{
    for (int i = blockIdx.x * blockDim.x + threadIdx.x; i < n;
         i += gridDim.x * blockDim.x) {
        float a = y[i];
        if (nt > 0) a += c1 * t1[i];
        if (nt > 1) a += c2 * t2[i];
        if (nt > 2) a += c3 * t3[i];
        if (nt > 3) a += c4 * t4[i];
        if (nt > 4) a += c5 * t5[i];
        out[i] = f2bf(a);
    }
}

// y += h*sum(Bi*ki); also emits bf16(y) as next step's stage-1 GEMM operand
__global__ void combine_y(float* __restrict__ y, __bf16* __restrict__ ybf,
                          const float* __restrict__ k1, const float* __restrict__ k3,
                          const float* __restrict__ k4, const float* __restrict__ k5,
                          const float* __restrict__ k6,
                          float c1, float c3, float c4, float c5, float c6, int n)
{
    for (int i = blockIdx.x * blockDim.x + threadIdx.x; i < n;
         i += gridDim.x * blockDim.x) {
        float v = y[i] + c1 * k1[i] + c3 * k3[i] + c4 * k4[i] + c5 * k5[i] + c6 * k6[i];
        y[i]   = v;
        ybf[i] = f2bf(v);
    }
}

__global__ void init_y(const float* __restrict__ x, float* __restrict__ y,
                       __bf16* __restrict__ ybf, int n)
{
    for (int i = blockIdx.x * blockDim.x + threadIdx.x; i < n;
         i += gridDim.x * blockDim.x) {
        float v = x[i];
        y[i]   = v;
        ybf[i] = f2bf(v);
    }
}

__global__ void cvt_bf16(const float* __restrict__ in, __bf16* __restrict__ out, int n)
{
    for (int i = blockIdx.x * blockDim.x + threadIdx.x; i < n;
         i += gridDim.x * blockDim.x) out[i] = f2bf(in[i]);
}

// ---------------------------------------------------------------------------
// Driver: 64 dopri5 steps, 6 stages each, 3 GEMMs per stage.
// ---------------------------------------------------------------------------
extern "C" void kernel_launch(void* const* d_in, const int* in_sizes, int n_in,
                              void* d_out, int out_size, void* d_ws, size_t ws_size,
                              hipStream_t stream)
{
    (void)in_sizes; (void)n_in; (void)out_size; (void)ws_size;

    const float* x  = (const float*)d_in[0];
    const float* W1 = (const float*)d_in[1];
    const float* b1 = (const float*)d_in[2];
    const float* W2 = (const float*)d_in[3];
    const float* b2 = (const float*)d_in[4];
    const float* W3 = (const float*)d_in[5];
    const float* b3 = (const float*)d_in[6];

    const int Bt = 4096, D = 512, H = 1024;
    float* y = (float*)d_out;           // y lives in d_out, updated in place

    char* ws = (char*)d_ws;
    size_t off = 0;
    auto carve = [&](size_t bytes) -> void* {
        void* p = ws + off;
        off += (bytes + 255) & ~(size_t)255;
        return p;
    };
    __bf16* W1b  = (__bf16*)carve((size_t)H * D * 2);
    __bf16* W2b  = (__bf16*)carve((size_t)H * H * 2);
    __bf16* W3b  = (__bf16*)carve((size_t)D * H * 2);
    __bf16* ytmp = (__bf16*)carve((size_t)Bt * D * 2);
    __bf16* h1   = (__bf16*)carve((size_t)Bt * H * 2);
    __bf16* h2   = (__bf16*)carve((size_t)Bt * H * 2);
    float*  kbuf[6];
    for (int i = 0; i < 6; ++i) kbuf[i] = (float*)carve((size_t)Bt * D * 4);

    // one-time per launch: weights -> bf16, y = x (and ytmp = bf16(x))
    cvt_bf16<<<2048, 256, 0, stream>>>(W1, W1b, H * D);
    cvt_bf16<<<2048, 256, 0, stream>>>(W2, W2b, H * H);
    cvt_bf16<<<2048, 256, 0, stream>>>(W3, W3b, D * H);
    init_y  <<<2048, 256, 0, stream>>>(x, y, ytmp, Bt * D);

    const double hh = 1.0 / 64.0;
    const double Atab[6][5] = {
        {0, 0, 0, 0, 0},
        {1.0/5.0, 0, 0, 0, 0},
        {3.0/40.0, 9.0/40.0, 0, 0, 0},
        {44.0/45.0, -56.0/15.0, 32.0/9.0, 0, 0},
        {19372.0/6561.0, -25360.0/2187.0, 64448.0/6561.0, -212.0/729.0, 0},
        {9017.0/3168.0, -355.0/33.0, 46732.0/5247.0, 49.0/176.0, -5103.0/18656.0},
    };
    const float cB1 = (float)(hh * 35.0/384.0);
    const float cB3 = (float)(hh * 500.0/1113.0);
    const float cB4 = (float)(hh * 125.0/192.0);
    const float cB5 = (float)(hh * -2187.0/6784.0);
    const float cB6 = (float)(hh * 11.0/84.0);

    const dim3 blk(256);
    const dim3 gH(Bt / 128, H / 256);   // (32, 4): BN=256 for N=1024
    const dim3 gD(Bt / 128, D / 128);   // (32, 4): BN=128 for N=512

    for (int s = 0; s < 64; ++s) {
        for (int st = 0; st < 6; ++st) {
            if (st > 0) {   // stage 0 operand (bf16(y)) comes from combine/init
                const float c1 = (float)(hh * Atab[st][0]);
                const float c2 = (float)(hh * Atab[st][1]);
                const float c3 = (float)(hh * Atab[st][2]);
                const float c4 = (float)(hh * Atab[st][3]);
                const float c5 = (float)(hh * Atab[st][4]);
                prep_stage<<<2048, 256, 0, stream>>>(
                    y,
                    kbuf[0], st > 1 ? kbuf[1] : y,
                    st > 2 ? kbuf[2] : y, st > 3 ? kbuf[3] : y,
                    st > 4 ? kbuf[4] : y,
                    c1, c2, c3, c4, c5, st, ytmp, Bt * D);
            }
            gemm_bf16_tn<1, 4><<<gH, blk, 0, stream>>>(ytmp, W1b, b1, h1, Bt, H, D);
            gemm_bf16_tn<1, 4><<<gH, blk, 0, stream>>>(h1,   W2b, b2, h2, Bt, H, H);
            gemm_bf16_tn<0, 2><<<gD, blk, 0, stream>>>(h2,   W3b, b3, kbuf[st], Bt, D, H);
        }
        combine_y<<<2048, 256, 0, stream>>>(y, ytmp, kbuf[0], kbuf[2], kbuf[3],
                                            kbuf[4], kbuf[5],
                                            cB1, cB3, cB4, cB5, cB6, Bt * D);
    }
}
